// _GalerkinAttentionLayer_3032246911130
// MI455X (gfx1250) — matrix operand order repaired and checked by reading, not compiled, not run on hardware
//
#include <hip/hip_runtime.h>

// ---------------------------------------------------------------------------
// Galerkin linear attention, fused bf16-WMMA pipeline for gfx1250 (MI455X).
// D = A*B+C via v_wmma_f32_16x16x32_bf16 (wave32). f32 accumulate everywhere,
// epilogues (RoPE / LayerNorm / mask / bias) in f32 registers.
// ---------------------------------------------------------------------------

typedef __bf16 bf16_t;
typedef __attribute__((ext_vector_type(16))) __bf16 v16bf;
typedef __attribute__((ext_vector_type(8)))  __bf16 v8bf;
typedef __attribute__((ext_vector_type(8)))  float  v8f;

#define HID   512
#define NHEAD 8
#define HD    64
#define SEQ   8192
#define NB    4
#define EPS   1e-5f
// log2(10000)/8  (RoPE: freq[i] = 10000^(-i/8), i = 0..15)
#define L2F   1.6609640474436813f

__device__ __forceinline__ v8bf ld8(const bf16_t* p) {
  return *(const v8bf*)p;
}
__device__ __forceinline__ v16bf mk16(v8bf lo, v8bf hi) {
  return __builtin_shufflevector(lo, hi, 0,1,2,3,4,5,6,7,8,9,10,11,12,13,14,15);
}
__device__ __forceinline__ v8f wmma_bf16(v16bf a, v16bf b, v8f c) {
  // 8 args: (neg_a, A, neg_b, B, c_mod, C, reuse_a, reuse_b)
  return __builtin_amdgcn_wmma_f32_16x16x32_bf16(false, a, false, b, (short)0, c,
                                                 false, false);
}

// ---------------------------------------------------------------------------
// proj_kernel<MODE>: C[128x128] tile of X[SEQ x 512] @ W[512 x 512], bf16 WMMA.
//   MODE 0 (K): RoPE + kv_mask + LayerNorm + kv_mask -> Knt[b][h][d][SEQ] bf16
//   MODE 1 (V):        kv_mask + LayerNorm + kv_mask -> Vnt[b][h][d][SEQ] bf16
//   MODE 2 (Q): RoPE + q_mask                        -> Qr [b][SEQ][512]  bf16
//   MODE 3 (O): + bias, * q_mask                     -> out[b][SEQ][512]  f32
// Block: 256 threads = 8 waves; wave (wm in 0..3, wn in 0..1) owns 32x64.
// ---------------------------------------------------------------------------
template<int MODE>
__global__ __launch_bounds__(256)
void proj_kernel(const float* __restrict__ X,
                 const float* __restrict__ W,
                 const float* __restrict__ coords,
                 const unsigned char* __restrict__ mask,
                 const float* __restrict__ lnw, const float* __restrict__ lnb,
                 const float* __restrict__ bias,
                 bf16_t* __restrict__ outT,   // MODE 0/1
                 bf16_t* __restrict__ outR,   // MODE 2
                 float*  __restrict__ outF)   // MODE 3
{
  __shared__ __align__(16) bf16_t sA[128][40];   // [m][k], k padded 32->40
  __shared__ __align__(16) bf16_t sB[128][40];   // [n][k] (W tile transposed)

  const int b    = blockIdx.z;
  const int s0   = blockIdx.x * 128;
  const int n0   = blockIdx.y * 128;
  const int tid  = threadIdx.x;
  const int lane = tid & 31;
  const int wave = tid >> 5;
  const int wm   = wave & 3;        // 4 row-blocks of 32
  const int wn   = wave >> 2;       // 2 col-blocks of 64 (one head for K/V/Q)
  const int lr   = lane & 15;
  const int half = lane >> 4;

  const float* Xb = X + (size_t)b * SEQ * HID;

  v8f acc[2][4] = {};

  for (int k0 = 0; k0 < HID; k0 += 32) {
    { // stage A tile 128x32 f32 -> bf16 (row-contiguous)
      const int r  = tid >> 1;
      const int kk = (tid & 1) * 16;
      const float4* src = (const float4*)(Xb + (size_t)(s0 + r) * HID + k0 + kk);
      #pragma unroll
      for (int q = 0; q < 4; ++q) {
        float4 f = src[q];
        sA[r][kk + q*4 + 0] = (bf16_t)f.x;
        sA[r][kk + q*4 + 1] = (bf16_t)f.y;
        sA[r][kk + q*4 + 2] = (bf16_t)f.z;
        sA[r][kk + q*4 + 3] = (bf16_t)f.w;
      }
      if (k0 + 32 < HID)
        __builtin_prefetch(Xb + (size_t)(s0 + r) * HID + k0 + 32 + kk, 0, 1);
    }
    { // stage B tile: W[k][n] -> sB[n][k] (transposed, f32 -> bf16)
      const int kr = tid >> 3;
      const int nc = (tid & 7) * 16;
      const float4* src = (const float4*)(W + (size_t)(k0 + kr) * HID + n0 + nc);
      #pragma unroll
      for (int q = 0; q < 4; ++q) {
        float4 f = src[q];
        sB[nc + q*4 + 0][kr] = (bf16_t)f.x;
        sB[nc + q*4 + 1][kr] = (bf16_t)f.y;
        sB[nc + q*4 + 2][kr] = (bf16_t)f.z;
        sB[nc + q*4 + 3][kr] = (bf16_t)f.w;
      }
      if (k0 + 32 < HID)
        __builtin_prefetch(W + (size_t)(k0 + 32 + kr) * HID + n0 + nc, 0, 1);
    }
    __syncthreads();

    // A frag: row = M (lane%16); 16 contiguous K at (lane/16)*16
    v16bf af[2];
    #pragma unroll
    for (int mt = 0; mt < 2; ++mt) {
      const bf16_t* pr = &sA[wm*32 + mt*16 + lr][half*16];
      af[mt] = mk16(ld8(pr), ld8(pr + 8));
    }
    // B frag: col = N (lane%16); 16 contiguous K at (lane/16)*16
    v16bf bfr[4];
    #pragma unroll
    for (int nt = 0; nt < 4; ++nt) {
      const bf16_t* pr = &sB[wn*64 + nt*16 + lr][half*16];
      bfr[nt] = mk16(ld8(pr), ld8(pr + 8));
    }
    #pragma unroll
    for (int mt = 0; mt < 2; ++mt)
      #pragma unroll
      for (int nt = 0; nt < 4; ++nt)
        acc[mt][nt] = wmma_bf16(af[mt], bfr[nt], acc[mt][nt]);
    __syncthreads();
  }

  // ---- epilogues (C layout: elem j of v8f -> M = j + 8*half, N = lr) ----
  if constexpr (MODE == 2) {               // Q: RoPE + q_mask, row-major bf16
    const float fi = __builtin_exp2f(-(float)lr * L2F);
    const float* cb = coords + (size_t)b * SEQ * 2;
    const unsigned char* mb = mask + (size_t)b * SEQ;
    #pragma unroll
    for (int mt = 0; mt < 2; ++mt)
      #pragma unroll
      for (int j = 0; j < 8; ++j) {
        const int s = s0 + wm*32 + mt*16 + half*8 + j;
        const float qm = mb[s] ? 1.0f : 0.0f;
        const float sx = __sinf(cb[2*s]   * fi), cx = __cosf(cb[2*s]   * fi);
        const float sy = __sinf(cb[2*s+1] * fi), cy = __cosf(cb[2*s+1] * fi);
        const float x1 = acc[mt][0][j], x2 = acc[mt][1][j];
        const float x3 = acc[mt][2][j], x4 = acc[mt][3][j];
        bf16_t* o = outR + ((size_t)b*SEQ + s) * HID + n0 + wn*64;
        o[ 0 + lr] = (bf16_t)((x1*cx - x2*sx) * qm);
        o[16 + lr] = (bf16_t)((x1*sx + x2*cx) * qm);
        o[32 + lr] = (bf16_t)((x3*cy - x4*sy) * qm);
        o[48 + lr] = (bf16_t)((x3*sy + x4*cy) * qm);
      }
  } else if constexpr (MODE == 3) {        // O: + bias, * q_mask, f32 out
    const unsigned char* mb = mask + (size_t)b * SEQ;
    float bi[4];
    #pragma unroll
    for (int nt = 0; nt < 4; ++nt) bi[nt] = bias[n0 + wn*64 + nt*16 + lr];
    #pragma unroll
    for (int mt = 0; mt < 2; ++mt)
      #pragma unroll
      for (int j = 0; j < 8; ++j) {
        const int s = s0 + wm*32 + mt*16 + half*8 + j;
        const float qm = mb[s] ? 1.0f : 0.0f;
        float* o = outF + ((size_t)b*SEQ + s) * HID + n0 + wn*64;
        #pragma unroll
        for (int nt = 0; nt < 4; ++nt)
          o[nt*16 + lr] = (acc[mt][nt][j] + bi[nt]) * qm;
      }
  } else {  // K/V: (RoPE) + mask + LayerNorm + mask, transposed bf16 out
    __shared__ __align__(16) bf16_t sOut[128][136];  // [feat][s], padded
    const float fi = __builtin_exp2f(-(float)lr * L2F);
    const float* cb = coords + (size_t)b * SEQ * 2;
    const unsigned char* mb = mask + (size_t)b * SEQ;
    float w_[4], b_[4];
    #pragma unroll
    for (int nt = 0; nt < 4; ++nt) {
      w_[nt] = lnw[nt*16 + lr]; b_[nt] = lnb[nt*16 + lr];
    }
    #pragma unroll
    for (int mt = 0; mt < 2; ++mt)
      #pragma unroll
      for (int j = 0; j < 8; ++j) {
        const int sl = wm*32 + mt*16 + half*8 + j;
        const int s  = s0 + sl;
        const float kvm = mb[s] ? 1.0f : 0.0f;
        float v0, v1, v2, v3;
        if constexpr (MODE == 0) {
          const float sx = __sinf(cb[2*s]   * fi), cx = __cosf(cb[2*s]   * fi);
          const float sy = __sinf(cb[2*s+1] * fi), cy = __cosf(cb[2*s+1] * fi);
          const float x1 = acc[mt][0][j], x2 = acc[mt][1][j];
          const float x3 = acc[mt][2][j], x4 = acc[mt][3][j];
          v0 = (x1*cx - x2*sx) * kvm;  v1 = (x1*sx + x2*cx) * kvm;
          v2 = (x3*cy - x4*sy) * kvm;  v3 = (x3*sy + x4*cy) * kvm;
        } else {
          v0 = acc[mt][0][j]*kvm; v1 = acc[mt][1][j]*kvm;
          v2 = acc[mt][2][j]*kvm; v3 = acc[mt][3][j]*kvm;
        }
        // LayerNorm over the head's 64 dims: 4 tiles/lane + 16-lane xor-reduce
        float sm = v0 + v1 + v2 + v3;
        float sq = v0*v0 + v1*v1 + v2*v2 + v3*v3;
        #pragma unroll
        for (int d = 1; d < 16; d <<= 1) {
          sm += __shfl_xor(sm, d, 32);
          sq += __shfl_xor(sq, d, 32);
        }
        const float mean = sm * (1.0f/64.0f);
        const float var  = sq * (1.0f/64.0f) - mean*mean;
        const float rstd = rsqrtf(var + EPS);
        v0 = ((v0-mean)*rstd*w_[0] + b_[0]) * kvm;
        v1 = ((v1-mean)*rstd*w_[1] + b_[1]) * kvm;
        v2 = ((v2-mean)*rstd*w_[2] + b_[2]) * kvm;
        v3 = ((v3-mean)*rstd*w_[3] + b_[3]) * kvm;
        sOut[wn*64 +  0 + lr][sl] = (bf16_t)v0;
        sOut[wn*64 + 16 + lr][sl] = (bf16_t)v1;
        sOut[wn*64 + 32 + lr][sl] = (bf16_t)v2;
        sOut[wn*64 + 48 + lr][sl] = (bf16_t)v3;
      }
    __syncthreads();
    // coalesced feature-major store: outT[b][h][d][SEQ]
    const int f  = tid >> 1;
    const int sh = tid & 1;
    const int g  = n0 + f;
    const int h  = g >> 6, d = g & 63;
    bf16_t* dst = outT + (((size_t)(b*NHEAD + h))*HD + d) * SEQ + s0 + sh*64;
    const uint4* ps = (const uint4*)&sOut[f][sh*64];
    uint4* pd = (uint4*)dst;
    #pragma unroll
    for (int q = 0; q < 8; ++q) pd[q] = ps[q];
  }
}

// ---------------------------------------------------------------------------
// kv_outer: KV^T[b][h][e][d] = (1/n) * sum_s Kn[s][d] * Vn[s][e], bf16 out.
// One workgroup per (b,h); each wave accumulates a 1024-row s-slice into a
// full 64x64 f32 tile set (16 WMMAs/step), cross-wave reduce via ds_add_f32.
// ---------------------------------------------------------------------------
__global__ __launch_bounds__(256)
void kv_outer_kernel(const bf16_t* __restrict__ Knt,
                     const bf16_t* __restrict__ Vnt,
                     const unsigned char* __restrict__ maskKV,
                     bf16_t* __restrict__ KVm)   // [B][H][64(e)][64(d)]
{
  __shared__ float accS[64 * 65];   // padded: bank = (d + e) % 64
  __shared__ float redS[8];
  __shared__ float nInvS;

  const int bh = blockIdx.x;
  const int b  = bh >> 3;
  const int tid = threadIdx.x, lane = tid & 31, wave = tid >> 5;
  const int lr = lane & 15, half = lane >> 4;

  // n = max(1, sum kv_mask[b]);  also zero the LDS accumulator
  unsigned int cnt = 0;
  const unsigned char* mb = maskKV + (size_t)b * SEQ;
  for (int i = tid; i < SEQ; i += 256) cnt += mb[i] ? 1u : 0u;
  #pragma unroll
  for (int d = 1; d < 32; d <<= 1) cnt += __shfl_xor(cnt, d, 32);
  if (lane == 0) redS[wave] = (float)cnt;
  for (int i = tid; i < 64 * 65; i += 256) accS[i] = 0.0f;
  __syncthreads();
  if (tid == 0) {
    float n = 0.f;
    for (int i = 0; i < 8; ++i) n += redS[i];
    nInvS = 1.0f / fmaxf(n, 1.0f);
  }

  const bf16_t* Kb = Knt + (size_t)bh * HD * SEQ;
  const bf16_t* Vb = Vnt + (size_t)bh * HD * SEQ;
  v8f acc[4][4] = {};
  const int sw0 = wave * 1024;
  for (int ks = 0; ks < 32; ++ks) {
    const int sb = sw0 + ks * 32;
    v16bf af[4], bfr[4];
    #pragma unroll
    for (int dt = 0; dt < 4; ++dt) {   // A = Kn^T : row d, K = s contiguous
      const bf16_t* p = Kb + (size_t)(dt*16 + lr) * SEQ + sb + half*16;
      af[dt] = mk16(ld8(p), ld8(p + 8));
    }
    #pragma unroll
    for (int et = 0; et < 4; ++et) {   // B = Vn : col e, K = s contiguous
      const bf16_t* p = Vb + (size_t)(et*16 + lr) * SEQ + sb + half*16;
      bfr[et] = mk16(ld8(p), ld8(p + 8));
    }
    #pragma unroll
    for (int dt = 0; dt < 4; ++dt)
      #pragma unroll
      for (int et = 0; et < 4; ++et)
        acc[dt][et] = wmma_bf16(af[dt], bfr[et], acc[dt][et]);
  }
  #pragma unroll
  for (int dt = 0; dt < 4; ++dt)
    #pragma unroll
    for (int et = 0; et < 4; ++et)
      #pragma unroll
      for (int j = 0; j < 8; ++j) {
        const int d = dt*16 + half*8 + j;   // C: M = d
        const int e = et*16 + lr;           // C: N = e
        atomicAdd(&accS[d*65 + e], acc[dt][et][j]);
      }
  __syncthreads();
  const float nInv = nInvS;
  for (int i = tid; i < 4096; i += 256) {
    const int e = i >> 6, d = i & 63;
    KVm[(size_t)bh * 4096 + i] = (bf16_t)(accS[d*65 + e] * nInv);  // [e][d]
  }
}

// ---------------------------------------------------------------------------
// attn: attn[b][s][h*64+e] = sum_d Qr[b][s][h*64+d] * KV^T[b][h][e][d].
// Fragments loaded straight from global (KV is 8 KB/head, cache-resident).
// ---------------------------------------------------------------------------
__global__ __launch_bounds__(256)
void attn_kernel(const bf16_t* __restrict__ Qr,
                 const bf16_t* __restrict__ KVm,
                 float* __restrict__ attn)
{
  const int b  = blockIdx.z;
  const int s0 = blockIdx.x * 128;
  const int hq = blockIdx.y;          // heads hq*4 .. hq*4+3
  const int tid = threadIdx.x, lane = tid & 31, wave = tid >> 5;
  const int wm = wave & 3, wn = wave >> 2;
  const int lr = lane & 15, half = lane >> 4;

  const bf16_t* Qb  = Qr  + (size_t)b * SEQ * HID;
  const bf16_t* KVb = KVm + (size_t)b * NHEAD * 4096;

  v8f acc[2][8] = {};
  #pragma unroll
  for (int hh = 0; hh < 2; ++hh) {
    const int h = hq*4 + wn*2 + hh;
    #pragma unroll
    for (int ks = 0; ks < 2; ++ks) {
      v16bf af[2];
      #pragma unroll
      for (int mt = 0; mt < 2; ++mt) {
        const bf16_t* p = Qb + (size_t)(s0 + wm*32 + mt*16 + lr) * HID
                             + h*64 + ks*32 + half*16;
        af[mt] = mk16(ld8(p), ld8(p + 8));
      }
      v16bf bfr[4];
      #pragma unroll
      for (int et = 0; et < 4; ++et) {
        const bf16_t* p = KVb + h*4096 + (et*16 + lr)*64 + ks*32 + half*16;
        bfr[et] = mk16(ld8(p), ld8(p + 8));
      }
      #pragma unroll
      for (int mt = 0; mt < 2; ++mt)
        #pragma unroll
        for (int et = 0; et < 4; ++et)
          acc[mt][hh*4 + et] = wmma_bf16(af[mt], bfr[et], acc[mt][hh*4 + et]);
    }
  }
  #pragma unroll
  for (int mt = 0; mt < 2; ++mt)
    #pragma unroll
    for (int hh = 0; hh < 2; ++hh)
      #pragma unroll
      for (int et = 0; et < 4; ++et)
        #pragma unroll
        for (int j = 0; j < 8; ++j) {
          const int s = s0 + wm*32 + mt*16 + half*8 + j;
          attn[((size_t)b*SEQ + s)*HID + (hq*4 + wn*2 + hh)*64 + et*16 + lr] =
              acc[mt][hh*4 + et][j];
        }
}

// ---------------------------------------------------------------------------
extern "C" void kernel_launch(void* const* d_in, const int* in_sizes, int n_in,
                              void* d_out, int out_size, void* d_ws, size_t ws_size,
                              hipStream_t stream) {
  (void)in_sizes; (void)n_in; (void)out_size; (void)ws_size;
  const float* q_feat    = (const float*)d_in[0];
  const float* kv_feat   = (const float*)d_in[1];
  const float* q_coords  = (const float*)d_in[2];
  const float* kv_coords = (const float*)d_in[3];
  const unsigned char* kv_mask = (const unsigned char*)d_in[4];
  const unsigned char* q_mask  = (const unsigned char*)d_in[5];
  const float* Wq = (const float*)d_in[6];
  const float* Wk = (const float*)d_in[7];
  const float* Wv = (const float*)d_in[8];
  const float* Wo = (const float*)d_in[9];
  const float* bo = (const float*)d_in[10];
  const float* k_w = (const float*)d_in[11];
  const float* k_b = (const float*)d_in[12];
  const float* v_w = (const float*)d_in[13];
  const float* v_b = (const float*)d_in[14];

  // workspace layout (96.25 MiB total):
  //   [0, 32M)   Knt bf16 [B][H][64][SEQ]
  //   [32M,64M)  Vnt bf16
  //   [64M,96M)  Qr  bf16 [B][SEQ][512]
  //   [96M, +256K) KVm bf16 [B][H][64][64]
  //   attn f32 [B][SEQ][512] aliases [0,64M) — safe: kv_outer (reader of
  //   Knt/Vnt) completes before attn_kernel writes, stream-ordered.
  char* ws = (char*)d_ws;
  const size_t KVT_BYTES = (size_t)NB * NHEAD * HD * SEQ * sizeof(bf16_t);
  bf16_t* Knt = (bf16_t*)(ws);
  bf16_t* Vnt = (bf16_t*)(ws + KVT_BYTES);
  bf16_t* Qr  = (bf16_t*)(ws + 2 * KVT_BYTES);
  bf16_t* KVm = (bf16_t*)(ws + 3 * KVT_BYTES);
  float*  attn = (float*)(ws);

  dim3 blk(256);
  dim3 gproj(SEQ / 128, HID / 128, NB);

  proj_kernel<0><<<gproj, blk, 0, stream>>>(kv_feat, Wk, kv_coords, kv_mask,
                                            k_w, k_b, nullptr, Knt, nullptr, nullptr);
  proj_kernel<1><<<gproj, blk, 0, stream>>>(kv_feat, Wv, kv_coords, kv_mask,
                                            v_w, v_b, nullptr, Vnt, nullptr, nullptr);
  proj_kernel<2><<<gproj, blk, 0, stream>>>(q_feat, Wq, q_coords, q_mask,
                                            nullptr, nullptr, nullptr,
                                            nullptr, Qr, nullptr);
  kv_outer_kernel<<<dim3(NB * NHEAD), blk, 0, stream>>>(Knt, Vnt, kv_mask, KVm);
  attn_kernel<<<dim3(SEQ / 128, 2, NB), blk, 0, stream>>>(Qr, KVm, attn);
  proj_kernel<3><<<gproj, blk, 0, stream>>>(attn, Wo, nullptr, q_mask,
                                            nullptr, nullptr, bo,
                                            nullptr, nullptr, (float*)d_out);
}
